// SplitNN_7095285973799
// MI455X (gfx1250) — compile-verified, matrix operand order checked by reading, not run on hardware
//
#include <hip/hip_runtime.h>
#include <math.h>

typedef __attribute__((ext_vector_type(16))) _Float16 v16h;
typedef __attribute__((ext_vector_type(8)))  _Float16 v8h;
typedef __attribute__((ext_vector_type(8)))  float    v8f;

#define LCA_ITERS 500
#define ETA       (1.0f / 1000.0f)
#define LAMBDA    0.5f

// ---------------------------------------------------------------------------
// Per-sample standardization: x = (x - mean) / (std + 1e-8), over C*H*W.
// ---------------------------------------------------------------------------
__global__ void standardize_kernel(const float* __restrict__ in,
                                   float* __restrict__ out, int n) {
  __shared__ float s_sum[256], s_sq[256];
  const int b = blockIdx.x, t = threadIdx.x;
  const float* x = in + (size_t)b * n;
  float sum = 0.f, sq = 0.f;
  for (int i = t; i < n; i += 256) { float v = x[i]; sum += v; sq += v * v; }
  s_sum[t] = sum; s_sq[t] = sq; __syncthreads();
  for (int o = 128; o > 0; o >>= 1) {
    if (t < o) { s_sum[t] += s_sum[t + o]; s_sq[t] += s_sq[t + o]; }
    __syncthreads();
  }
  const float mean = s_sum[0] / (float)n;
  float var = s_sq[0] / (float)n - mean * mean;
  var = var > 0.f ? var : 0.f;
  const float inv = 1.f / (sqrtf(var) + 1e-8f);
  float* y = out + (size_t)b * n;
  for (int i = t; i < n; i += 256) y[i] = (x[i] - mean) * inv;
}

// ---------------------------------------------------------------------------
// BatchNorm2d with batch statistics (training math, eps=1e-5). One block/ch.
// ---------------------------------------------------------------------------
__global__ void bn_kernel(const float* __restrict__ in, float* __restrict__ out,
                          const float* __restrict__ gam, const float* __restrict__ bet,
                          int B, int C, int HW) {
  __shared__ float s_sum[256], s_sq[256];
  const int c = blockIdx.x, t = threadIdx.x;
  const int n = B * HW;
  float sum = 0.f, sq = 0.f;
  for (int i = t; i < n; i += 256) {
    int bi = i / HW, s = i % HW;
    float v = in[((size_t)bi * C + c) * HW + s];
    sum += v; sq += v * v;
  }
  s_sum[t] = sum; s_sq[t] = sq; __syncthreads();
  for (int o = 128; o > 0; o >>= 1) {
    if (t < o) { s_sum[t] += s_sum[t + o]; s_sq[t] += s_sq[t + o]; }
    __syncthreads();
  }
  const float mean = s_sum[0] / (float)n;
  float var = s_sq[0] / (float)n - mean * mean;
  var = var > 0.f ? var : 0.f;
  const float scale = rsqrtf(var + 1e-5f) * gam[c];
  const float shift = bet[c] - mean * scale;
  for (int i = t; i < n; i += 256) {
    int bi = i / HW, s = i % HW;
    size_t idx = ((size_t)bi * C + c) * HW + s;
    out[idx] = in[idx] * scale + shift;
  }
}

// ---------------------------------------------------------------------------
// Direct 5x5 conv, stride 1, pad 2 (LCA drives and conv1..3). Tiny workload.
// ---------------------------------------------------------------------------
__global__ void conv5_kernel(const float* __restrict__ in, const float* __restrict__ w,
                             const float* __restrict__ bias, float* __restrict__ out,
                             int B, int Cin, int Cout, int H, int W, int do_relu) {
  int idx = blockIdx.x * blockDim.x + threadIdx.x;
  int total = B * Cout * H * W;
  if (idx >= total) return;
  int x = idx % W, y = (idx / W) % H, o = (idx / (W * H)) % Cout, b = idx / (W * H * Cout);
  float acc = bias ? bias[o] : 0.f;
  for (int c = 0; c < Cin; ++c) {
    const float* ip = in + ((size_t)b * Cin + c) * H * W;
    const float* wp = w + ((size_t)o * Cin + c) * 25;
    for (int i = 0; i < 5; ++i) {
      int yy = y + i - 2; if (yy < 0 || yy >= H) continue;
      for (int j = 0; j < 5; ++j) {
        int xx = x + j - 2; if (xx < 0 || xx >= W) continue;
        acc += ip[yy * W + xx] * wp[i * 5 + j];
      }
    }
  }
  if (do_relu) acc = acc > 0.f ? acc : 0.f;
  out[idx] = acc;
}

// ---------------------------------------------------------------------------
// Gram: G[f,g,dy,dx] = conv(w, w, pad=4), emitted as f16 in WMMA-A friendly
// layout Gbuf[tap][f][k], K padded to 32 (zeros for k >= F -> exact).
// ---------------------------------------------------------------------------
__global__ void gram_kernel(const float* __restrict__ w, _Float16* __restrict__ g,
                            int F, int Cin) {
  int idx = blockIdx.x * blockDim.x + threadIdx.x;
  int total = 81 * F * 32;
  if (idx >= total) return;
  int k = idx & 31;
  int f = (idx >> 5) % F;
  int tap = idx / (32 * F);
  if (k >= F) { g[idx] = (_Float16)0.f; return; }
  int dy = tap / 9, dx = tap % 9;
  float acc = 0.f;
  for (int c = 0; c < Cin; ++c) {
    const float* wf = w + ((size_t)f * Cin + c) * 25;
    const float* wk = w + ((size_t)k * Cin + c) * 25;
    for (int p = 0; p < 5; ++p) {
      int ry = dy + p - 4; if (ry < 0 || ry >= 5) continue;
      for (int q = 0; q < 5; ++q) {
        int rx = dx + q - 4; if (rx < 0 || rx >= 5) continue;
        acc += wf[ry * 5 + rx] * wk[p * 5 + q];
      }
    }
  }
  g[idx] = (_Float16)acc;
}

// ---------------------------------------------------------------------------
// 2x2 maxpool, stride 2.
// ---------------------------------------------------------------------------
__global__ void maxpool2_kernel(const float* __restrict__ in, float* __restrict__ out,
                                int BC, int H, int W) {
  int OW = W >> 1, OH = H >> 1;
  int idx = blockIdx.x * blockDim.x + threadIdx.x;
  int total = BC * OH * OW;
  if (idx >= total) return;
  int ox = idx % OW, oy = (idx / OW) % OH, bc = idx / (OW * OH);
  const float* p = in + (size_t)bc * H * W + (oy * 2) * W + ox * 2;
  float m0 = p[0] > p[1] ? p[0] : p[1];
  float m1 = p[W] > p[W + 1] ? p[W] : p[W + 1];
  out[idx] = m0 > m1 ? m0 : m1;
}

// ---------------------------------------------------------------------------
// Fully-connected: y = x @ W^T + b (optional relu). Tiny.
// ---------------------------------------------------------------------------
__global__ void fc_kernel(const float* __restrict__ x, const float* __restrict__ w,
                          const float* __restrict__ b, float* __restrict__ y,
                          int B, int In, int Out, int do_relu) {
  int idx = blockIdx.x * blockDim.x + threadIdx.x;
  int total = B * Out;
  if (idx >= total) return;
  int o = idx % Out, bi = idx / Out;
  float acc = b[o];
  const float* xp = x + (size_t)bi * In;
  const float* wp = w + (size_t)o * In;
  for (int i = 0; i < In; ++i) acc += xp[i] * wp[i];
  if (do_relu) acc = acc > 0.f ? acc : 0.f;
  y[idx] = acc;
}

// ---------------------------------------------------------------------------
// LCA iteration kernel. grid = B images, 1024 threads = 32 waves; wave w owns
// image row y = w, split into two 16-pixel groups (WMMA N dim). Membrane u
// and drive live in registers in WMMA-D layout (lane<16: N=lane, VGPR j ->
// M=j; lane>=16 -> M=j+8). Sparse code a (f16, 4-px zero halo, channel-
// contiguous 64B/pixel) and Gram (f16 [tap][cout][cin], K padded to 32) live
// in LDS. Each 9x9 tap: load Gram A-tiles once (shared by both pixel
// groups), then one v_wmma_f32_16x16x32_f16 per (group, tile) with f32
// accumulation chained across all 81 taps. All register arrays are indexed
// with compile-time constants (groups/tiles fully unrolled) so nothing
// lowers to scratch / v_movrels. Two barriers per iteration; the 500-step
// loop never touches HBM.
// ---------------------------------------------------------------------------
template <int F>
__global__ void __launch_bounds__(1024)
lca_kernel(const float* __restrict__ drive, const _Float16* __restrict__ G,
           float* __restrict__ out) {
  constexpr int TILES = F / 16;
  constexpr int AH = 40;             // 32 + 2*4 halo
  constexpr int APIX = AH * AH;
  constexpr int GSZ = 81 * F * 32;

  extern __shared__ _Float16 smem[];
  _Float16* aS = smem;               // [APIX][32] f16 activations (padded)
  _Float16* gS = smem + APIX * 32;   // [81][F][32] f16 gram

  const int tid  = threadIdx.x;
  const int lane = tid & 31;
  const int wave = tid >> 5;         // row y
  const int img  = blockIdx.x;
  const int col  = lane & 15;        // pixel within group (WMMA N)
  const int half = lane >> 4;        // K sub-range / M+8 selector
  const int kb   = half * 8;
  const int y    = wave;

  for (int i = tid; i < APIX * 32; i += 1024) aS[i] = (_Float16)0.f;
  for (int i = tid; i < GSZ; i += 1024) gS[i] = G[i];

  v8f u[2][TILES];
  v8f dr[2][TILES];
#pragma unroll
  for (int grp = 0; grp < 2; ++grp) {
    const int p = y * 32 + grp * 16 + col;
#pragma unroll
    for (int t = 0; t < TILES; ++t) {
#pragma unroll
      for (int j = 0; j < 8; ++j) {
        const int c = t * 16 + kb + j;
        dr[grp][t][j] = drive[((size_t)img * F + c) * 1024 + p];
        u[grp][t][j]  = 0.f;
      }
    }
  }
  __syncthreads();

  const v8f vzero = {0.f, 0.f, 0.f, 0.f, 0.f, 0.f, 0.f, 0.f};

#pragma unroll 1
  for (int iter = 0; iter < LCA_ITERS; ++iter) {
    // ---- phase 1: a = relu(u - lambda) -> LDS (f16, padded layout) ----
#pragma unroll
    for (int grp = 0; grp < 2; ++grp) {
      const int pp = ((y + 4) * AH + (grp * 16 + col + 4)) * 32;
#pragma unroll
      for (int t = 0; t < TILES; ++t) {
#pragma unroll
        for (int j = 0; j < 8; ++j) {
          float a = u[grp][t][j] - LAMBDA;
          a = a > 0.f ? a : 0.f;
          aS[pp + t * 16 + kb + j] = (_Float16)a;
        }
      }
    }
    __syncthreads();

    // ---- phase 2: inhib = sum over 81 taps of G_tap @ a_shift ----
    v8f acc[2][TILES];
#pragma unroll
    for (int grp = 0; grp < 2; ++grp)
#pragma unroll
      for (int t = 0; t < TILES; ++t) acc[grp][t] = vzero;

#pragma unroll 1
    for (int i = 0; i < 9; ++i) {
#pragma unroll
      for (int jj = 0; jj < 9; ++jj) {
        const int tap = i * 9 + jj;
        // A operands: G[tap][t*16 + row][cin 0..31]; shared by both groups
        union { v16h v; v8h h[2]; } Am[TILES];
#pragma unroll
        for (int t = 0; t < TILES; ++t) {
          const _Float16* ap = &gS[((size_t)tap * F + (t * 16 + col)) * 32 + kb];
          Am[t].h[0] = *(const v8h*)ap;         // K = kb .. kb+7
          Am[t].h[1] = *(const v8h*)(ap + 16);  // K = kb+16 .. kb+23
        }
#pragma unroll
        for (int grp = 0; grp < 2; ++grp) {
          // B operand: a[cin 0..31][16 px], column split across lane halves
          const _Float16* bp =
              &aS[((y + i) * AH + (grp * 16 + col + jj)) * 32 + kb];
          union { v16h v; v8h h[2]; } Bm;
          Bm.h[0] = *(const v8h*)bp;
          Bm.h[1] = *(const v8h*)(bp + 16);
#pragma unroll
          for (int t = 0; t < TILES; ++t) {
            acc[grp][t] = __builtin_amdgcn_wmma_f32_16x16x32_f16(
                false, Am[t].v, false, Bm.v, (short)0, acc[grp][t], false, false);
          }
        }
      }
    }

    // ---- phase 3: u += eta * (drive - u - inhib + a) ----
#pragma unroll
    for (int grp = 0; grp < 2; ++grp) {
#pragma unroll
      for (int t = 0; t < TILES; ++t) {
#pragma unroll
        for (int j = 0; j < 8; ++j) {
          float uu = u[grp][t][j];
          float a = uu - LAMBDA; a = a > 0.f ? a : 0.f;
          u[grp][t][j] = uu + ETA * (dr[grp][t][j] - uu - acc[grp][t][j] + a);
        }
      }
    }
    __syncthreads();
  }

  // final sparse code a = relu(u - lambda) -> global [img][F][32*32]
#pragma unroll
  for (int grp = 0; grp < 2; ++grp) {
    const int p = y * 32 + grp * 16 + col;
#pragma unroll
    for (int t = 0; t < TILES; ++t) {
#pragma unroll
      for (int j = 0; j < 8; ++j) {
        const int c = t * 16 + kb + j;
        float a = u[grp][t][j] - LAMBDA;
        out[((size_t)img * F + c) * 1024 + p] = a > 0.f ? a : 0.f;
      }
    }
  }
}

// ---------------------------------------------------------------------------
// Host-side orchestration.
// ---------------------------------------------------------------------------
static inline int cdiv(int a, int b) { return (a + b - 1) / b; }

extern "C" void kernel_launch(void* const* d_in, const int* in_sizes, int n_in,
                              void* d_out, int out_size, void* d_ws, size_t ws_size,
                              hipStream_t stream) {
  (void)in_sizes; (void)n_in; (void)out_size; (void)ws_size;

  const float* x      = (const float*)d_in[0];
  const float* w_lca1 = (const float*)d_in[1];
  const float* w_lca2 = (const float*)d_in[2];
  const float* bn1_g  = (const float*)d_in[3];
  const float* bn1_b  = (const float*)d_in[4];
  const float* bn2_g  = (const float*)d_in[5];
  const float* bn2_b  = (const float*)d_in[6];
  const float* w_c1   = (const float*)d_in[7];
  const float* b_c1   = (const float*)d_in[8];
  const float* bn3_g  = (const float*)d_in[9];
  const float* bn3_b  = (const float*)d_in[10];
  const float* w_c2   = (const float*)d_in[11];
  const float* b_c2   = (const float*)d_in[12];
  const float* bn4_g  = (const float*)d_in[13];
  const float* bn4_b  = (const float*)d_in[14];
  const float* w_c3   = (const float*)d_in[15];
  const float* b_c3   = (const float*)d_in[16];
  const float* bn5_g  = (const float*)d_in[17];
  const float* bn5_b  = (const float*)d_in[18];
  const float* w_fc1  = (const float*)d_in[19];
  const float* b_fc1  = (const float*)d_in[20];
  const float* w_fc2  = (const float*)d_in[21];
  const float* b_fc2  = (const float*)d_in[22];

  char* ws = (char*)d_ws;
  size_t off = 0;
  auto alloc = [&](size_t bytes) -> void* {
    void* p = ws + off;
    off = (off + bytes + 255) & ~(size_t)255;
    return p;
  };

  const int B = 4;
  float*    xn     = (float*)   alloc((size_t)B * 3 * 1024 * 4);
  float*    drive1 = (float*)   alloc((size_t)B * 16 * 1024 * 4);
  _Float16* G1     = (_Float16*)alloc((size_t)81 * 16 * 32 * 2);
  float*    h1     = (float*)   alloc((size_t)B * 16 * 1024 * 4);
  float*    h1b    = (float*)   alloc((size_t)B * 16 * 1024 * 4);
  float*    h1s    = (float*)   alloc((size_t)B * 16 * 1024 * 4);
  float*    drive2 = (float*)   alloc((size_t)B * 32 * 1024 * 4);
  _Float16* G2     = (_Float16*)alloc((size_t)81 * 32 * 32 * 2);
  float*    h2     = (float*)   alloc((size_t)B * 32 * 1024 * 4);
  float*    p2     = (float*)   alloc((size_t)B * 32 * 256 * 4);
  float*    b2o    = (float*)   alloc((size_t)B * 32 * 256 * 4);
  float*    c1o    = (float*)   alloc((size_t)B * 64 * 256 * 4);
  float*    p3     = (float*)   alloc((size_t)B * 64 * 64 * 4);
  float*    b3o    = (float*)   alloc((size_t)B * 64 * 64 * 4);
  float*    c2o    = (float*)   alloc((size_t)B * 128 * 64 * 4);
  float*    p4     = (float*)   alloc((size_t)B * 128 * 16 * 4);
  float*    b4o    = (float*)   alloc((size_t)B * 128 * 16 * 4);
  float*    c3o    = (float*)   alloc((size_t)B * 256 * 16 * 4);
  float*    p5     = (float*)   alloc((size_t)B * 256 * 4 * 4);
  float*    b5o    = (float*)   alloc((size_t)B * 256 * 4 * 4);
  float*    f1o    = (float*)   alloc((size_t)B * 512 * 4);

  const size_t smem1 = ((size_t)40 * 40 * 32 + (size_t)81 * 16 * 32) * 2; // 185344 B
  const size_t smem2 = ((size_t)40 * 40 * 32 + (size_t)81 * 32 * 32) * 2; // 268288 B

  // ---- LCA stage 1 ----
  standardize_kernel<<<B, 256, 0, stream>>>(x, xn, 3 * 1024);
  {
    int total = B * 16 * 1024;
    conv5_kernel<<<cdiv(total, 256), 256, 0, stream>>>(xn, w_lca1, nullptr, drive1,
                                                       B, 3, 16, 32, 32, 0);
  }
  {
    int total = 81 * 16 * 32;
    gram_kernel<<<cdiv(total, 256), 256, 0, stream>>>(w_lca1, G1, 16, 3);
  }
  lca_kernel<16><<<B, 1024, smem1, stream>>>(drive1, G1, h1);
  bn_kernel<<<16, 256, 0, stream>>>(h1, h1b, bn1_g, bn1_b, B, 16, 1024);

  // ---- LCA stage 2 ----
  standardize_kernel<<<B, 256, 0, stream>>>(h1b, h1s, 16 * 1024);
  {
    int total = B * 32 * 1024;
    conv5_kernel<<<cdiv(total, 256), 256, 0, stream>>>(h1s, w_lca2, nullptr, drive2,
                                                       B, 16, 32, 32, 32, 0);
  }
  {
    int total = 81 * 32 * 32;
    gram_kernel<<<cdiv(total, 256), 256, 0, stream>>>(w_lca2, G2, 32, 16);
  }
  lca_kernel<32><<<B, 1024, smem2, stream>>>(drive2, G2, h2);
  maxpool2_kernel<<<cdiv(B * 32 * 256, 256), 256, 0, stream>>>(h2, p2, B * 32, 32, 32);
  bn_kernel<<<32, 256, 0, stream>>>(p2, b2o, bn2_g, bn2_b, B, 32, 256);

  // ---- conv tail ----
  conv5_kernel<<<cdiv(B * 64 * 256, 256), 256, 0, stream>>>(b2o, w_c1, b_c1, c1o,
                                                            B, 32, 64, 16, 16, 1);
  maxpool2_kernel<<<cdiv(B * 64 * 64, 256), 256, 0, stream>>>(c1o, p3, B * 64, 16, 16);
  bn_kernel<<<64, 256, 0, stream>>>(p3, b3o, bn3_g, bn3_b, B, 64, 64);

  conv5_kernel<<<cdiv(B * 128 * 64, 256), 256, 0, stream>>>(b3o, w_c2, b_c2, c2o,
                                                            B, 64, 128, 8, 8, 1);
  maxpool2_kernel<<<cdiv(B * 128 * 16, 256), 256, 0, stream>>>(c2o, p4, B * 128, 8, 8);
  bn_kernel<<<128, 256, 0, stream>>>(p4, b4o, bn4_g, bn4_b, B, 128, 16);

  conv5_kernel<<<cdiv(B * 256 * 16, 256), 256, 0, stream>>>(b4o, w_c3, b_c3, c3o,
                                                            B, 128, 256, 4, 4, 1);
  maxpool2_kernel<<<cdiv(B * 256 * 4, 256), 256, 0, stream>>>(c3o, p5, B * 256, 4, 4);
  bn_kernel<<<256, 256, 0, stream>>>(p5, b5o, bn5_g, bn5_b, B, 256, 4);

  // ---- FC head ----
  fc_kernel<<<cdiv(B * 512, 256), 256, 0, stream>>>(b5o, w_fc1, b_fc1, f1o,
                                                    B, 1024, 512, 1);
  fc_kernel<<<1, 256, 0, stream>>>(f1o, w_fc2, b_fc2, (float*)d_out,
                                   B, 512, 10, 0);
}
